// EventImage_Fusion_Block_48765058679324
// MI455X (gfx1250) — compile-verified, matrix operand order checked by reading, not run on hardware
//
#include <hip/hip_runtime.h>
#include <hip/hip_bf16.h>
#include <math.h>

typedef __attribute__((ext_vector_type(16))) _Float16 v16h;
typedef __attribute__((ext_vector_type(8)))  _Float16 h8;
typedef __attribute__((ext_vector_type(8)))  float    v8f;

#define NPIX   16384
#define CB     256
#define BATCH  8
#define HEADSN 8
#define CHD    32
#define HIDDEN 512
#define MROWS  (BATCH*NPIX)   // 131072

__device__ __forceinline__ float gelu_exact(float x) {
  return 0.5f * x * (1.0f + erff(x * 0.70710678118654752440f));
}

// CDNA5 async copy: global memory -> LDS, 16 bytes per lane, tracked by ASYNCcnt.
__device__ __forceinline__ void async_copy_b128(unsigned lds_off, unsigned long long gaddr) {
  asm volatile("global_load_async_to_lds_b128 %0, %1, off"
               :: "v"(lds_off), "v"(gaddr) : "memory");
}
__device__ __forceinline__ void async_wait0() {
  asm volatile("s_wait_asynccnt 0x0" ::: "memory");
}

// Load a 16x32 f16 WMMA fragment from a row-major LDS tile [rows][32].
// Per CDNA5 ISA 7.12.2: lanes 0-15 hold row M=lane, K pairs {0..7,16..23};
// lanes 16-31 hold row M=lane-16, K pairs {8..15,24..31}. Each half is one
// contiguous 16-byte LDS read. Same striping applies to B (row = N column).
__device__ __forceinline__ v16h lds_frag(const _Float16* tile, int row0, int lane) {
  int r = row0 + (lane & 15);
  int hsel = lane >> 4;
  const h8* p = (const h8*)(tile + r * 32 + hsel * 8);
  h8 lo = p[0];   // k = 8*hsel .. 8*hsel+7
  h8 hi = p[2];   // k = 16 + 8*hsel ..
  v16h f;
  #pragma unroll
  for (int i = 0; i < 8; ++i) { f[i] = lo[i]; f[i + 8] = hi[i]; }
  return f;
}

__global__ void k_f32_to_f16(const float* __restrict__ src, _Float16* __restrict__ dst, int n) {
  int i = blockIdx.x * 256 + threadIdx.x;
  if (i < n) dst[i] = (_Float16)src[i];
}

__global__ void k_zero(float* __restrict__ p, int n) {
  int i = blockIdx.x * 256 + threadIdx.x;
  if (i < n) p[i] = 0.0f;
}

// LayerNorm over channels of [B,C,H*W]; output f16 row-major [row, C].
// Single strided read pass (values cached in LDS as f16), then a cooperative
// fully-coalesced output pass: each wave writes one contiguous 512B row.
// Block = 64 pixels (2 waves). LDS: 256*65*2 + stats = ~34 KB (of 320 KB/WGP).
__global__ void k_ln_spatial(const float* __restrict__ src, const float* __restrict__ w,
                             const float* __restrict__ b, _Float16* __restrict__ dst) {
  __shared__ _Float16 cache[CB * 65];   // [c][pixel] padded stride 65 vs 64
  __shared__ float mu_s[64], inv_s[64];
  const int t = threadIdx.x;            // 0..63
  const int rowBase = blockIdx.x * 64;
  const int row = rowBase + t;
  const int bb  = row >> 14;
  const int pix = row & (NPIX - 1);
  const float* base = src + (size_t)bb * CB * NPIX + pix;
  float s = 0.f, ss = 0.f;
  for (int c = 0; c < CB; ++c) {
    float v = base[(size_t)c * NPIX];   // coalesced: lanes = consecutive pixels
    s += v; ss += v * v;
    cache[c * 65 + t] = (_Float16)v;
  }
  float mu  = s  * (1.0f / CB);
  float var = ss * (1.0f / CB) - mu * mu;
  mu_s[t]  = mu;
  inv_s[t] = rsqrtf(var + 1e-5f);
  __syncthreads();
  // 64 rows x 32 chunks(8 halves): wave-contiguous stores.
  #pragma unroll 4
  for (int i = 0; i < 32; ++i) {
    int chunk = t + i * 64;             // 0..2047
    int r  = chunk >> 5;                // pixel within tile
    int cc = chunk & 31;                // 8-half group
    float m = mu_s[r], iv = inv_s[r];
    h8 o;
    #pragma unroll
    for (int j = 0; j < 8; ++j) {
      int c = cc * 8 + j;
      float v = (float)cache[c * 65 + r];
      o[j] = (_Float16)((v - m) * iv * w[c] + b[c]);
    }
    *(h8*)(dst + (size_t)(rowBase + r) * CB + cc * 8) = o;
  }
}

// Generic WMMA GEMM: C[m,n] = sum_k A[m,k] * W[n,k]  (weights stored (out,in))
// Block tile 128x128, 8 waves (4x2), wave tile 32x64 (2x4 accumulators), K step 32.
// Tiles staged with GLOBAL_LOAD_ASYNC_TO_LDS_B128 (ASYNCcnt), no VGPR round-trip.
// mode 0: store f16              (QKV projections)
// mode 1: + image residual (spatial layout), store f32 "fused"
// mode 2: + bias, GELU, store f16 (FC1)
// mode 3: + bias, GELU, + fusedIn, store f32 to spatial [B,C,H*W] (FC2 final)
__global__ void k_gemm_wmma(const _Float16* __restrict__ A, const _Float16* __restrict__ W,
                            int K, int N, int mode,
                            const float* __restrict__ bias,
                            const float* __restrict__ resImage,
                            const float* __restrict__ fusedIn,
                            _Float16* __restrict__ out16,
                            float* __restrict__ outf) {
  __shared__ __align__(16) _Float16 As[128 * 32];
  __shared__ __align__(16) _Float16 Bs[128 * 32];
  const int t    = threadIdx.x;
  const int lane = t & 31;
  const int wave = t >> 5;
  const int wm   = wave >> 1;                 // 0..3
  const int wn   = wave & 1;                  // 0..1
  const int rowBase = blockIdx.x * 128;
  const int colBase = blockIdx.y * 128;

  v8f acc[2][4];
  #pragma unroll
  for (int mi = 0; mi < 2; ++mi)
    #pragma unroll
    for (int ni = 0; ni < 4; ++ni) acc[mi][ni] = {};

  for (int k0 = 0; k0 < K; k0 += 32) {
    #pragma unroll
    for (int i = 0; i < 2; ++i) {
      int ci = t + i * 256;                  // 512 16B chunks per 128x32 tile
      int r = ci >> 2, cc = ci & 3;
      const _Float16* ga = A + (size_t)(rowBase + r) * K + k0 + cc * 8;
      const _Float16* gb = W + (size_t)(colBase + r) * K + k0 + cc * 8;
      async_copy_b128((unsigned)(unsigned long long)&As[r * 32 + cc * 8],
                      (unsigned long long)ga);
      async_copy_b128((unsigned)(unsigned long long)&Bs[r * 32 + cc * 8],
                      (unsigned long long)gb);
      if (k0 + 32 < K) {
        __builtin_prefetch((const void*)(ga + 32), 0, 1);
        __builtin_prefetch((const void*)(gb + 32), 0, 1);
      }
    }
    async_wait0();
    __syncthreads();
    v16h af[2], bf[4];
    #pragma unroll
    for (int mi = 0; mi < 2; ++mi) af[mi] = lds_frag(As, wm * 32 + mi * 16, lane);
    #pragma unroll
    for (int ni = 0; ni < 4; ++ni) bf[ni] = lds_frag(Bs, wn * 64 + ni * 16, lane);
    #pragma unroll
    for (int mi = 0; mi < 2; ++mi)
      #pragma unroll
      for (int ni = 0; ni < 4; ++ni)
        acc[mi][ni] = __builtin_amdgcn_wmma_f32_16x16x32_f16(
            false, af[mi], false, bf[ni], (short)0, acc[mi][ni], false, false);
    __syncthreads();
  }

  // Epilogue: C/D layout — VGPR vr holds M=vr (lanes 0-15) / M=vr+8 (lanes 16-31), N=lane&15.
  const int rsub = 8 * (lane >> 4);
  const int csub = lane & 15;
  #pragma unroll
  for (int mi = 0; mi < 2; ++mi) {
    #pragma unroll
    for (int ni = 0; ni < 4; ++ni) {
      #pragma unroll
      for (int vr = 0; vr < 8; ++vr) {
        int gRow = rowBase + wm * 32 + mi * 16 + vr + rsub;
        int gCol = colBase + wn * 64 + ni * 16 + csub;
        float val = acc[mi][ni][vr];
        if (mode == 0) {
          out16[(size_t)gRow * N + gCol] = (_Float16)val;
        } else if (mode == 1) {
          int bb = gRow >> 14, pix = gRow & (NPIX - 1);
          val += resImage[((size_t)bb * CB + gCol) * NPIX + pix];
          outf[(size_t)gRow * N + gCol] = val;
        } else if (mode == 2) {
          val = gelu_exact(val + bias[gCol]);
          out16[(size_t)gRow * N + gCol] = (_Float16)val;
        } else {
          val = gelu_exact(val + bias[gCol]);
          val += fusedIn[(size_t)gRow * CB + gCol];
          int bb = gRow >> 14, pix = gRow & (NPIX - 1);
          outf[((size_t)bb * CB + gCol) * NPIX + pix] = val;
        }
      }
    }
  }
}

// Column sums of squares of q,k per (batch, channel) -> atomic accumulate.
__global__ void k_colsumsq(const _Float16* __restrict__ q, const _Float16* __restrict__ k,
                           float* __restrict__ nq, float* __restrict__ nk) {
  int col = threadIdx.x;
  int rowBase = blockIdx.x * 128;
  int bb = rowBase >> 14;
  float aq = 0.f, ak = 0.f;
  for (int r = 0; r < 128; ++r) {
    size_t idx = (size_t)(rowBase + r) * CB + col;
    float qv = (float)q[idx]; aq += qv * qv;
    float kv = (float)k[idx]; ak += kv * kv;
  }
  atomicAdd(&nq[bb * CB + col], aq);
  atomicAdd(&nk[bb * CB + col], ak);
}

// Per-(b,h) Gram matrix S[32,32] = q·k^T over N=16384 pixels via WMMA.
// K reduction split into 16 slices (grid.y) for occupancy; f32 atomic reduce.
// 4 waves per block, each owning one 16x16 quadrant; transpose-on-LDS-write.
__global__ void k_gram_wmma(const _Float16* __restrict__ q, const _Float16* __restrict__ k,
                            float* __restrict__ S) {
  __shared__ __align__(16) _Float16 qT[32 * 32];
  __shared__ __align__(16) _Float16 kT[32 * 32];
  int blk  = blockIdx.x;             // b*8 + h
  int bb   = blk >> 3, h = blk & 7;
  int nBeg = blockIdx.y * (NPIX / 16);
  int nEnd = nBeg + (NPIX / 16);
  int t    = threadIdx.x;            // 0..127
  int lane = t & 31;
  int wave = t >> 5;                 // 0..3
  int mi   = wave >> 1, nj = wave & 1;
  int d    = t & 31;
  int kg   = t >> 5;                 // 0..3
  size_t base = (size_t)bb * NPIX * CB + h * CHD;
  v8f acc = {};
  for (int n0 = nBeg; n0 < nEnd; n0 += 32) {
    #pragma unroll
    for (int i = 0; i < 8; ++i) {
      int kk = kg * 8 + i;
      size_t gi = base + (size_t)(n0 + kk) * CB + d;   // coalesced over d
      qT[d * 32 + kk] = q[gi];
      kT[d * 32 + kk] = k[gi];
    }
    __syncthreads();
    v16h af = lds_frag(qT, mi * 16, lane);
    v16h bf = lds_frag(kT, nj * 16, lane);
    acc = __builtin_amdgcn_wmma_f32_16x16x32_f16(false, af, false, bf, (short)0, acc, false, false);
    __syncthreads();
  }
  int rsub = 8 * (lane >> 4), csub = lane & 15;
  #pragma unroll
  for (int vr = 0; vr < 8; ++vr) {
    int row = mi * 16 + vr + rsub;
    int col = nj * 16 + csub;
    atomicAdd(&S[(size_t)blk * (CHD * CHD) + row * CHD + col], acc[vr]);
  }
}

// Softmax over the 32-wide rows with L2-norm scales folded in.
__global__ void k_softmax(const float* __restrict__ S, const float* __restrict__ nq,
                          const float* __restrict__ nk, const float* __restrict__ temp,
                          _Float16* __restrict__ attn) {
  int blk = blockIdx.x;              // b*8 + h
  int bb  = blk >> 3, h = blk & 7;
  int ci  = threadIdx.x;             // 0..31
  __shared__ float invk[32];
  float tq = temp[h];
  invk[ci] = 1.0f / fmaxf(sqrtf(nk[bb * CB + h * CHD + ci]), 1e-12f);
  __syncthreads();
  float iq = 1.0f / fmaxf(sqrtf(nq[bb * CB + h * CHD + ci]), 1e-12f);
  const float* Sr = S + (size_t)blk * (CHD * CHD) + ci * CHD;
  float mx = -1e30f;
  for (int j = 0; j < CHD; ++j) mx = fmaxf(mx, Sr[j] * iq * invk[j] * tq);
  float sum = 0.f;
  for (int j = 0; j < CHD; ++j) sum += __expf(Sr[j] * iq * invk[j] * tq - mx);
  float is = 1.0f / sum;
  _Float16* Ar = attn + (size_t)blk * (CHD * CHD) + ci * CHD;
  for (int j = 0; j < CHD; ++j) Ar[j] = (_Float16)(__expf(Sr[j] * iq * invk[j] * tq - mx) * is);
}

// Block-diagonal o[row, h*32+n] = sum_k v[row, h*32+k] * attn[b,h,n,k] via WMMA.
__global__ void k_headattn_wmma(const _Float16* __restrict__ v, const _Float16* __restrict__ attn,
                                _Float16* __restrict__ o) {
  __shared__ __align__(16) _Float16 Att[32 * 32];
  __shared__ __align__(16) _Float16 Vs[128 * 32];
  int t    = threadIdx.x;
  int lane = t & 31, wave = t >> 5;   // 8 waves, one 16-row slab each
  int rowBase = blockIdx.x * 128;
  int bb = rowBase >> 14;
  for (int h = 0; h < HEADSN; ++h) {
    #pragma unroll
    for (int i = 0; i < 4; ++i) {
      int idx = t + i * 256;
      Att[idx] = attn[(size_t)(bb * 8 + h) * (CHD * CHD) + idx];
    }
    #pragma unroll
    for (int i = 0; i < 2; ++i) {
      int ci = t + i * 256;
      int r = ci >> 2, cc = ci & 3;
      const h8* gv = (const h8*)(v + (size_t)(rowBase + r) * CB + h * CHD + cc * 8);
      *(h8*)(&Vs[r * 32 + cc * 8]) = *gv;
    }
    __syncthreads();
    v16h af = lds_frag(Vs, wave * 16, lane);
    v16h b0 = lds_frag(Att, 0, lane);
    v16h b1 = lds_frag(Att, 16, lane);
    v8f z = {};
    v8f a0 = __builtin_amdgcn_wmma_f32_16x16x32_f16(false, af, false, b0, (short)0, z, false, false);
    v8f a1 = __builtin_amdgcn_wmma_f32_16x16x32_f16(false, af, false, b1, (short)0, z, false, false);
    int rsub = 8 * (lane >> 4), csub = lane & 15;
    #pragma unroll
    for (int vr = 0; vr < 8; ++vr) {
      int gRow = rowBase + wave * 16 + vr + rsub;
      o[(size_t)gRow * CB + h * CHD + csub]      = (_Float16)a0[vr];
      o[(size_t)gRow * CB + h * CHD + 16 + csub] = (_Float16)a1[vr];
    }
    __syncthreads();
  }
}

// LayerNorm over contiguous f32 rows -> f16; one wave (32 lanes) per row.
__global__ void k_ln_rows(const float* __restrict__ src, const float* __restrict__ w,
                          const float* __restrict__ b, _Float16* __restrict__ dst) {
  int lane = threadIdx.x & 31;
  int wave = threadIdx.x >> 5;
  int row  = blockIdx.x * 8 + wave;
  const float* rp = src + (size_t)row * CB;
  float v[8]; float s = 0.f, ss = 0.f;
  #pragma unroll
  for (int i = 0; i < 8; ++i) { v[i] = rp[i * 32 + lane]; s += v[i]; ss += v[i] * v[i]; }
  #pragma unroll
  for (int off = 16; off > 0; off >>= 1) {
    s  += __shfl_xor(s,  off, 32);
    ss += __shfl_xor(ss, off, 32);
  }
  float mu  = s  * (1.f / CB);
  float var = ss * (1.f / CB) - mu * mu;
  float inv = rsqrtf(var + 1e-5f);
  _Float16* op = dst + (size_t)row * CB;
  #pragma unroll
  for (int i = 0; i < 8; ++i) {
    int c = i * 32 + lane;
    op[c] = (_Float16)((v[i] - mu) * inv * w[c] + b[c]);
  }
}

extern "C" void kernel_launch(void* const* d_in, const int* in_sizes, int n_in,
                              void* d_out, int out_size, void* d_ws, size_t ws_size,
                              hipStream_t stream) {
  (void)in_sizes; (void)n_in; (void)out_size; (void)ws_size;
  const float* image       = (const float*)d_in[0];
  const float* event       = (const float*)d_in[1];
  const float* ln_img_w    = (const float*)d_in[2];
  const float* ln_img_b    = (const float*)d_in[3];
  const float* ln_evt_w    = (const float*)d_in[4];
  const float* ln_evt_b    = (const float*)d_in[5];
  const float* Wq          = (const float*)d_in[6];
  const float* Wk          = (const float*)d_in[7];
  const float* Wv          = (const float*)d_in[8];
  const float* Wo          = (const float*)d_in[9];
  const float* temperature = (const float*)d_in[10];
  const float* norm2_w     = (const float*)d_in[11];
  const float* norm2_b     = (const float*)d_in[12];
  const float* fc1_w       = (const float*)d_in[13];
  const float* fc1_b       = (const float*)d_in[14];
  const float* fc2_w       = (const float*)d_in[15];
  const float* fc2_b       = (const float*)d_in[16];

  char* w = (char*)d_ws;
  const size_t M     = (size_t)MROWS;
  const size_t szA16 = M * CB * 2;                 // 64 MiB per f16 activation
  _Float16* xn   = (_Float16*)(w);
  _Float16* yn   = (_Float16*)(w + szA16);
  _Float16* q16  = (_Float16*)(w + 2 * szA16);
  _Float16* k16  = (_Float16*)(w + 3 * szA16);
  _Float16* v16p = (_Float16*)(w + 4 * szA16);
  float*    fused = (float*)(w + 5 * szA16);       // f32 [M,256]
  _Float16* h16  = (_Float16*)(w + 5 * szA16 + M * CB * 4);  // f16 [M,512]
  char* sm = w + 5 * szA16 + M * CB * 4 + M * HIDDEN * 2;
  _Float16* wq16  = (_Float16*)sm; sm += CB * CB * 2;
  _Float16* wk16  = (_Float16*)sm; sm += CB * CB * 2;
  _Float16* wv16  = (_Float16*)sm; sm += CB * CB * 2;
  _Float16* wo16  = (_Float16*)sm; sm += CB * CB * 2;
  _Float16* f1w16 = (_Float16*)sm; sm += HIDDEN * CB * 2;
  _Float16* f2w16 = (_Float16*)sm; sm += CB * HIDDEN * 2;
  float*    nq    = (float*)sm;    sm += BATCH * CB * 4;
  float*    nk    = (float*)sm;    sm += BATCH * CB * 4;
  float*    S     = (float*)sm;    sm += BATCH * HEADSN * CHD * CHD * 4;
  _Float16* attn16 = (_Float16*)sm; sm += BATCH * HEADSN * CHD * CHD * 2;
  _Float16* o16 = xn;   // reuse: xn dead after Q projection
  _Float16* zln = yn;   // reuse: yn dead after K/V projections

  // 1) weights -> f16
  k_f32_to_f16<<<(CB * CB + 255) / 256, 256, 0, stream>>>(Wq, wq16, CB * CB);
  k_f32_to_f16<<<(CB * CB + 255) / 256, 256, 0, stream>>>(Wk, wk16, CB * CB);
  k_f32_to_f16<<<(CB * CB + 255) / 256, 256, 0, stream>>>(Wv, wv16, CB * CB);
  k_f32_to_f16<<<(CB * CB + 255) / 256, 256, 0, stream>>>(Wo, wo16, CB * CB);
  k_f32_to_f16<<<(HIDDEN * CB + 255) / 256, 256, 0, stream>>>(fc1_w, f1w16, HIDDEN * CB);
  k_f32_to_f16<<<(CB * HIDDEN + 255) / 256, 256, 0, stream>>>(fc2_w, f2w16, CB * HIDDEN);

  // 2) LayerNorms of the spatial inputs (single global read pass, LDS-cached)
  k_ln_spatial<<<M / 64, 64, 0, stream>>>(image, ln_img_w, ln_img_b, xn);
  k_ln_spatial<<<M / 64, 64, 0, stream>>>(event, ln_evt_w, ln_evt_b, yn);

  // 3) Q/K/V projections (WMMA, async-LDS staging)
  dim3 g256(M / 128, CB / 128);
  k_gemm_wmma<<<g256, 256, 0, stream>>>(xn, wq16, CB, CB, 0, nullptr, nullptr, nullptr, q16, nullptr);
  k_gemm_wmma<<<g256, 256, 0, stream>>>(yn, wk16, CB, CB, 0, nullptr, nullptr, nullptr, k16, nullptr);
  k_gemm_wmma<<<g256, 256, 0, stream>>>(yn, wv16, CB, CB, 0, nullptr, nullptr, nullptr, v16p, nullptr);

  // 4) zero the accumulators (nq, nk, S are contiguous), then norms
  k_zero<<<(2 * BATCH * CB + BATCH * HEADSN * CHD * CHD + 255) / 256, 256, 0, stream>>>(
      nq, 2 * BATCH * CB + BATCH * HEADSN * CHD * CHD);
  k_colsumsq<<<M / 128, 256, 0, stream>>>(q16, k16, nq, nk);

  // 5) Gram matrices (16-way K-split, atomic reduce) + softmax
  k_gram_wmma<<<dim3(BATCH * HEADSN, 16), 128, 0, stream>>>(q16, k16, S);
  k_softmax<<<BATCH * HEADSN, 32, 0, stream>>>(S, nq, nk, temperature, attn16);

  // 6) out = attn @ v (block-diagonal per head, WMMA)
  k_headattn_wmma<<<M / 128, 256, 0, stream>>>(v16p, attn16, o16);

  // 7) O projection + image residual -> fused (f32)
  k_gemm_wmma<<<g256, 256, 0, stream>>>(o16, wo16, CB, CB, 1, nullptr, image, nullptr, nullptr, fused);

  // 8) FFN: pre-norm, fc1+GELU, fc2+GELU, residual, spatial store to d_out
  k_ln_rows<<<M / 8, 256, 0, stream>>>(fused, norm2_w, norm2_b, zln);
  dim3 g512(M / 128, HIDDEN / 128);
  k_gemm_wmma<<<g512, 256, 0, stream>>>(zln, f1w16, CB, HIDDEN, 2, fc1_b, nullptr, nullptr, h16, nullptr);
  k_gemm_wmma<<<g256, 256, 0, stream>>>(h16, f2w16, HIDDEN, CB, 3, fc2_b, nullptr, fused, nullptr, (float*)d_out);
}
